// Attention_15350213116218
// MI455X (gfx1250) — matrix operand order repaired and checked by reading, not compiled, not run on hardware
//
#include <hip/hip_runtime.h>
#include <hip/hip_bf16.h>

typedef __bf16 bf16;
typedef __attribute__((ext_vector_type(16))) __bf16 v16bf;
typedef __attribute__((ext_vector_type(8)))  __bf16 v8bf;
typedef __attribute__((ext_vector_type(8)))  float  v8f;
typedef unsigned int u32x4 __attribute__((ext_vector_type(4)));
typedef int          i32x4 __attribute__((ext_vector_type(4)));
typedef int          i32x8 __attribute__((ext_vector_type(8)));

#define S_LEN  2048
#define DMODEL 2048
#define NH     32
#define NKV    8
#define HD     64
#define KV_D   (NKV * HD)   // 512

__device__ inline v8f wmma_bf16(v16bf a, v16bf b, v8f c) {
    return __builtin_amdgcn_wmma_f32_16x16x32_bf16(
        /*neg_a=*/false, a, /*neg_b=*/false, b,
        /*c_mod=*/(short)0, c, /*reuse_a=*/false, /*reuse_b=*/false);
}

__device__ inline v8f zero8() {
    v8f z;
    #pragma unroll
    for (int i = 0; i < 8; ++i) z[i] = 0.0f;
    return z;
}

union FragU { v16bf v; v8bf h[2]; };

// ---- TDM: DMA a 2-D bf16 tile (tile_w x tile_h, row stride in elems) into LDS ----
// D# packing per CDNA5 ISA §8.3/8.4 (2-byte elements, 2-D tile, groups 2/3 zero).
__device__ inline void tdm_load_tile_2d(const bf16* gsrc, bf16* ldsdst,
                                        unsigned tile_w, unsigned tile_h,
                                        unsigned row_stride) {
    unsigned lds32 = (unsigned)(unsigned long long)ldsdst;
    unsigned long long ga = (unsigned long long)gsrc;
    const unsigned td0 = row_stride;     // tensor dim0 extent (elems)
    const unsigned td1 = 1u << 20;       // tensor dim1 extent (rows) - generous

    u32x4 g0;
    g0.x = 1u;                                            // count=1 user descriptor
    g0.y = lds32;                                         // lds_addr (bytes)
    g0.z = (unsigned)(ga & 0xffffffffull);                // global_addr[31:0]
    g0.w = (unsigned)((ga >> 32) & 0x01ffffffull)         // global_addr[56:32]
         | (2u << 30);                                    // type=2 ("image")

    i32x8 g1;
    g1[0] = (int)(1u << 16);                              // wg_mask=0, data_size=1 (2B)
    g1[1] = (int)((td0 & 0xffffu) << 16);                 // tensor_dim0[15:0]
    g1[2] = (int)((td0 >> 16) | ((td1 & 0xffffu) << 16)); // td0[31:16] | td1[15:0]
    g1[3] = (int)((td1 >> 16) | (tile_w << 16));          // td1[31:16] | tile_dim0
    g1[4] = (int)tile_h;                                  // tile_dim1 | tile_dim2=0
    g1[5] = (int)row_stride;                              // tensor_dim0_stride[31:0]
    g1[6] = 0;                                            // stride[47:32] | dim1_stride lo
    g1[7] = 0;

    i32x4 g2 = {0, 0, 0, 0};
    i32x4 g3 = {0, 0, 0, 0};
#if defined(__clang_major__) && (__clang_major__ >= 23)
    i32x8 g4 = {0, 0, 0, 0, 0, 0, 0, 0};
    __builtin_amdgcn_tensor_load_to_lds(g0, g1, g2, g3, g4, 0);
#else
    __builtin_amdgcn_tensor_load_to_lds(g0, g1, g2, g3, 0);
#endif
}

// ---- per-lane async global->LDS 16B copy (ASYNCcnt tracked) ----
__device__ inline void async_copy_b128(const void* gsrc, void* ldsdst) {
    unsigned ldsoff = (unsigned)(unsigned long long)ldsdst;
    asm volatile("global_load_async_to_lds_b128 %0, %1, off"
                 :: "v"(ldsoff), "v"(gsrc) : "memory");
}
__device__ inline void wait_async0() {
    asm volatile("s_wait_asynccnt 0x0" ::: "memory");
}

// ---------------- elementwise f32 -> bf16 ----------------
__global__ void cvt_f32_bf16(const float* __restrict__ in, bf16* __restrict__ out, int n) {
    int i = blockIdx.x * blockDim.x + threadIdx.x;
    int stride = gridDim.x * blockDim.x;
    for (; i < n; i += stride) out[i] = (bf16)in[i];
}

// ---------------- RoPE: f32 in -> bf16 out ----------------
__global__ void rope_kernel(const float* __restrict__ in, const float* __restrict__ fc,
                            const float* __restrict__ fs, bf16* __restrict__ outb,
                            int nheads, int pitch) {
    int idx = blockIdx.x * blockDim.x + threadIdx.x;
    int i = idx & 31;          // rotary pair index 0..31
    int t = idx >> 5;
    int h = t % nheads;
    int s = t / nheads;
    float c  = fc[(s << 5) + i];
    float sn = fs[(s << 5) + i];
    size_t base = (size_t)s * pitch + h * HD + (i << 1);
    float xr = in[base], xi = in[base + 1];
    outb[base]     = (bf16)(xr * c - xi * sn);
    outb[base + 1] = (bf16)(xr * sn + xi * c);
}

// ---------------- tiled bf16 GEMM, f32 out ----------------
// C[M,N] = A[M,K] @ B[K,N]; M,N mult of 128, K mult of 32.
// A tile staged by TDM (tensor_load_to_lds); B^T staged by VALU (transpose).
// Fragments preloaded into registers so the 8 WMMAs issue back-to-back.
__global__ __launch_bounds__(256)
void gemm_bf16_f32(const bf16* __restrict__ A, const bf16* __restrict__ B,
                   float* __restrict__ C, int M, int N, int K) {
    __shared__ __align__(64) bf16 ldsA[128 * 32];   // [row][k]   8 KB
    __shared__ __align__(64) bf16 ldsBT[128 * 32];  // [n][k]     8 KB (B transposed)

    const int tid  = threadIdx.x;
    const int lane = tid & 31;
    const int w    = tid >> 5;
    const int m0   = blockIdx.y * 128;
    const int n0   = blockIdx.x * 128;

    v8f acc[8];
    #pragma unroll
    for (int t = 0; t < 8; ++t) acc[t] = zero8();

    const int cidx  = lane & 15;
    const int kb8   = (lane >> 4) << 3;   // K sub-offset (0/8), unused
    const int kb16  = (lane >> 4) << 4;   // B-frag K sub-offset (0/16)

    for (int k0 = 0; k0 < K; k0 += 32) {
        __syncthreads();
        // stage A tile 128x32 via Tensor Data Mover (wave 0 issues the DMA)
        if (w == 0) {
            tdm_load_tile_2d(A + (size_t)m0 * K + k0, ldsA, 32u, 128u, (unsigned)K);
        }
        { // stage B^T: ldsBT[n][k] = B[(k0+k)*N + n0+n]
            int n  = tid >> 1;
            int kh = (tid & 1) << 4;
            const bf16* src = B + (size_t)(k0 + kh) * N + n0 + n;
            bf16 tmp[16];
            #pragma unroll
            for (int kk = 0; kk < 16; ++kk) tmp[kk] = src[(size_t)kk * N];
            #pragma unroll
            for (int kk = 0; kk < 16; ++kk) ldsBT[n * 32 + kh + kk] = tmp[kk];
        }
        if (w == 0) __builtin_amdgcn_s_wait_tensorcnt(0);
        __syncthreads();

        // preload A fragment + all 8 B fragments, then issue WMMAs back-to-back
        FragU af;
        int arow = (w << 4) + cidx;
        af.h[0] = *(const v8bf*)(&ldsA[arow * 32 + kb16]);
        af.h[1] = *(const v8bf*)(&ldsA[arow * 32 + 8 + kb16]);

        v16bf bfr[8];
        #pragma unroll
        for (int nt = 0; nt < 8; ++nt) {
            int brow = (nt << 4) + cidx;
            bfr[nt] = *(const v16bf*)(&ldsBT[brow * 32 + kb16]);
        }
        #pragma unroll
        for (int nt = 0; nt < 8; ++nt)
            acc[nt] = wmma_bf16(af.v, bfr[nt], acc[nt]);
    }

    // store C tiles (C layout: lane col = lane&15; rows (lane>>4)*8 + j)
    const int rbase = (lane >> 4) << 3;
    #pragma unroll
    for (int nt = 0; nt < 8; ++nt) {
        float* cp = C + (size_t)(m0 + (w << 4) + rbase) * N + n0 + (nt << 4) + cidx;
        #pragma unroll
        for (int j = 0; j < 8; ++j) cp[(size_t)j * N] = acc[nt][j];
    }
}

// ---------------- flash attention (GQA, causal) ----------------
// Grid: (S/128, NH). Block 256 threads = 8 waves; wave w handles queries q0+16w..+15.
__global__ __launch_bounds__(256)
void flash_attn(const bf16* __restrict__ Q, const bf16* __restrict__ Km,
                const bf16* __restrict__ Vm, bf16* __restrict__ Ctx) {
    __shared__ __align__(64) bf16 ldsK[32 * 64];    // [key][hd]  4 KB
    __shared__ __align__(64) bf16 ldsVT[64 * 32];   // [hd][key]  4 KB
    __shared__ __align__(64) bf16 ldsP[8][16 * 32]; // per-wave P 8 KB

    const int tid  = threadIdx.x;
    const int lane = tid & 31;
    const int w    = tid >> 5;
    const int q0   = blockIdx.x * 128;
    const int h    = blockIdx.y;
    const int kvh  = h >> 2;        // N_REP = 4
    const int qw   = q0 + (w << 4); // wave query base

    const int cidx  = lane & 15;
    const int kb8   = (lane >> 4) << 3;
    const int kb16  = (lane >> 4) << 4;
    const int rbase = (lane >> 4) << 3;

    // load Q fragments once (A layout, hd 0..31 and 32..63)
    FragU qf0, qf1;
    {
        const bf16* qp = Q + (size_t)(qw + cidx) * DMODEL + h * HD;
        qf0.h[0] = *(const v8bf*)(qp + kb16);
        qf0.h[1] = *(const v8bf*)(qp + 8 + kb16);
        qf1.h[0] = *(const v8bf*)(qp + 32 + kb16);
        qf1.h[1] = *(const v8bf*)(qp + 40 + kb16);
    }

    v8f O[4];
    #pragma unroll
    for (int t = 0; t < 4; ++t) O[t] = zero8();
    float mrow[8], lrow[8];
    #pragma unroll
    for (int j = 0; j < 8; ++j) { mrow[j] = -3.0e38f; lrow[j] = 0.0f; }

    const float scale = 0.125f; // 1/sqrt(64)
    const int nkb = (q0 + 128) / 32;

    for (int blk = 0; blk < nkb; ++blk) {
        const int kb = blk * 32;
        __syncthreads();
        { // stage K block asynchronously: ldsK[kk][hd] (no transpose -> ASYNC path)
            int kk  = tid >> 3;
            int hd0 = (tid & 7) << 3;
            async_copy_b128(Km + (size_t)(kb + kk) * KV_D + kvh * HD + hd0,
                            &ldsK[kk * 64 + hd0]);
        }
        { // stage V^T block (transpose, VALU path): ldsVT[hd][kk]
            int hd  = tid >> 2;
            int kk0 = (tid & 3) << 3;
            #pragma unroll
            for (int i2 = 0; i2 < 8; ++i2)
                ldsVT[hd * 32 + kk0 + i2] =
                    Vm[(size_t)(kb + kk0 + i2) * KV_D + kvh * HD + hd];
        }
        wait_async0();
        __syncthreads();

        if (kb <= qw + 15) {   // wave-uniform causal skip
            // scores: S = Q (16x64) @ K^T (64x32keys); preload 4 B-frags
            v16bf kf[4];
            kf[0] = *(const v16bf*)(&ldsK[cidx * 64 + kb16]);
            kf[1] = *(const v16bf*)(&ldsK[cidx * 64 + 32 + kb16]);
            kf[2] = *(const v16bf*)(&ldsK[(16 + cidx) * 64 + kb16]);
            kf[3] = *(const v16bf*)(&ldsK[(16 + cidx) * 64 + 32 + kb16]);
            v8f s0 = zero8(), s1 = zero8();
            s0 = wmma_bf16(qf0.v, kf[0], s0);
            s0 = wmma_bf16(qf1.v, kf[1], s0);
            s1 = wmma_bf16(qf0.v, kf[2], s1);
            s1 = wmma_bf16(qf1.v, kf[3], s1);

            float p0[8], p1[8], corr[8];
            #pragma unroll
            for (int j = 0; j < 8; ++j) {
                int qrow = qw + rbase + j;
                float a0 = s0[j] * scale; if (kb + cidx      > qrow) a0 = -3.0e38f;
                float a1 = s1[j] * scale; if (kb + 16 + cidx > qrow) a1 = -3.0e38f;
                float mx = fmaxf(a0, a1);
                mx = fmaxf(mx, __shfl_xor(mx, 1, 32));
                mx = fmaxf(mx, __shfl_xor(mx, 2, 32));
                mx = fmaxf(mx, __shfl_xor(mx, 4, 32));
                mx = fmaxf(mx, __shfl_xor(mx, 8, 32));
                float mn = fmaxf(mrow[j], mx);
                float e0 = __expf(a0 - mn);
                float e1 = __expf(a1 - mn);
                float cr = __expf(mrow[j] - mn);
                float rs = e0 + e1;
                rs += __shfl_xor(rs, 1, 32);
                rs += __shfl_xor(rs, 2, 32);
                rs += __shfl_xor(rs, 4, 32);
                rs += __shfl_xor(rs, 8, 32);
                lrow[j] = lrow[j] * cr + rs;
                mrow[j] = mn;
                corr[j] = cr;
                p0[j] = e0; p1[j] = e1;
            }
            #pragma unroll
            for (int t = 0; t < 4; ++t)
                #pragma unroll
                for (int j = 0; j < 8; ++j) O[t][j] *= corr[j];

            // route P through per-wave LDS: C-layout -> A-fragment layout
            bf16* pw = &ldsP[w][0];
            #pragma unroll
            for (int j = 0; j < 8; ++j) {
                pw[(rbase + j) * 32 + cidx]      = (bf16)p0[j];
                pw[(rbase + j) * 32 + 16 + cidx] = (bf16)p1[j];
            }
            FragU pf;
            pf.h[0] = *(const v8bf*)(&pw[cidx * 32 + kb16]);
            pf.h[1] = *(const v8bf*)(&pw[cidx * 32 + 8 + kb16]);

            // O += P (16x32) @ V (32x64); preload 4 V-frags, then 4 WMMAs
            v16bf vf[4];
            #pragma unroll
            for (int t = 0; t < 4; ++t)
                vf[t] = *(const v16bf*)(&ldsVT[((t << 4) + cidx) * 32 + kb16]);
            #pragma unroll
            for (int t = 0; t < 4; ++t)
                O[t] = wmma_bf16(pf.v, vf[t], O[t]);
        }
    }

    // normalize and store context (bf16)
    float inv[8];
    #pragma unroll
    for (int j = 0; j < 8; ++j) inv[j] = 1.0f / lrow[j];
    #pragma unroll
    for (int t = 0; t < 4; ++t) {
        #pragma unroll
        for (int j = 0; j < 8; ++j) {
            size_t idx = (size_t)(qw + rbase + j) * DMODEL + h * HD + (t << 4) + cidx;
            Ctx[idx] = (bf16)(O[t][j] * inv[j]);
        }
    }
}

extern "C" void kernel_launch(void* const* d_in, const int* in_sizes, int n_in,
                              void* d_out, int out_size, void* d_ws, size_t ws_size,
                              hipStream_t stream) {
    const float* x  = (const float*)d_in[0];
    const float* wq = (const float*)d_in[1];
    const float* wk = (const float*)d_in[2];
    const float* wv = (const float*)d_in[3];
    const float* wo = (const float*)d_in[4];
    const float* fc = (const float*)d_in[5];
    const float* fs = (const float*)d_in[6];
    float* out = (float*)d_out;

    const size_t SD  = (size_t)S_LEN * DMODEL; // 4M
    const size_t SKV = (size_t)S_LEN * KV_D;   // 1M

    char* ws = (char*)d_ws;
    size_t off = 0;
    auto carve = [&](size_t bytes) -> char* {
        char* p = ws + off;
        off += (bytes + 255) & ~(size_t)255;
        return p;
    };
    bf16*  xb   = (bf16*)carve(SD * 2);
    bf16*  wqb  = (bf16*)carve(SD * 2);
    bf16*  wkb  = (bf16*)carve(SKV * 2);
    bf16*  wvb  = (bf16*)carve(SKV * 2);
    bf16*  wob  = (bf16*)carve(SD * 2);
    float* qf   = (float*)carve(SD * 4);
    float* kf   = (float*)carve(SKV * 4);
    float* vf   = (float*)carve(SKV * 4);
    bf16*  qb   = (bf16*)carve(SD * 2);
    bf16*  kbuf = (bf16*)carve(SKV * 2);
    bf16*  vbuf = (bf16*)carve(SKV * 2);
    bf16*  ctx  = (bf16*)carve(SD * 2);
    (void)ws_size; (void)in_sizes; (void)n_in; (void)out_size;

    // 1. convert inputs to bf16
    cvt_f32_bf16<<<SD  / 256, 256, 0, stream>>>(x,  xb,  (int)SD);
    cvt_f32_bf16<<<SD  / 256, 256, 0, stream>>>(wq, wqb, (int)SD);
    cvt_f32_bf16<<<SKV / 256, 256, 0, stream>>>(wk, wkb, (int)SKV);
    cvt_f32_bf16<<<SKV / 256, 256, 0, stream>>>(wv, wvb, (int)SKV);
    cvt_f32_bf16<<<SD  / 256, 256, 0, stream>>>(wo, wob, (int)SD);

    // 2. QKV projections (WMMA GEMMs)
    gemm_bf16_f32<<<dim3(DMODEL / 128, S_LEN / 128), 256, 0, stream>>>(xb, wqb, qf, S_LEN, DMODEL, DMODEL);
    gemm_bf16_f32<<<dim3(KV_D   / 128, S_LEN / 128), 256, 0, stream>>>(xb, wkb, kf, S_LEN, KV_D,   DMODEL);
    gemm_bf16_f32<<<dim3(KV_D   / 128, S_LEN / 128), 256, 0, stream>>>(xb, wvb, vf, S_LEN, KV_D,   DMODEL);

    // 3. RoPE (f32 math) -> bf16; V convert
    rope_kernel<<<(S_LEN * NH  * (HD / 2)) / 256, 256, 0, stream>>>(qf, fc, fs, qb,   NH,  DMODEL);
    rope_kernel<<<(S_LEN * NKV * (HD / 2)) / 256, 256, 0, stream>>>(kf, fc, fs, kbuf, NKV, KV_D);
    cvt_f32_bf16<<<SKV / 256, 256, 0, stream>>>(vf, vbuf, (int)SKV);

    // 4. flash attention (causal, GQA)
    flash_attn<<<dim3(S_LEN / 128, NH), 256, 0, stream>>>(qb, kbuf, vbuf, ctx);

    // 5. output projection -> f32 result
    gemm_bf16_f32<<<dim3(DMODEL / 128, S_LEN / 128), 256, 0, stream>>>(ctx, wob, out, S_LEN, DMODEL, DMODEL);
}
